// BlockPathApproximators_6622839571383
// MI455X (gfx1250) — matrix operand order repaired and checked by reading, not compiled, not run on hardware
//
#include <hip/hip_runtime.h>
#include <hip/hip_bf16.h>

typedef __attribute__((ext_vector_type(16))) __bf16 v16bf;
typedef __attribute__((ext_vector_type(8)))  float  v8f;

constexpr int DIM      = 1024;
constexpr int RANK     = 128;
constexpr int NEXP     = 7;
constexpr int FULL_KEY = 7;     // TOTAL_KEYS - 1: no approximator
constexpr int NTOK     = 8 * 4096;
constexpr int CAP      = NTOK;  // per-expert bucket capacity

// ---------------- workspace layout (bytes) ----------------
constexpr size_t OFF_COUNTS = 0;                                    // 8 ints
constexpr size_t OFF_PERM   = 256;                                  // NEXP*CAP ints
constexpr size_t OFF_WDB    = OFF_PERM + (size_t)NEXP * CAP * 4;    // Wd bf16
constexpr size_t OFF_WUB    = OFF_WDB + (size_t)NEXP * RANK * DIM * 2;
constexpr size_t OFF_DOWN   = OFF_WUB + (size_t)NEXP * DIM * RANK * 2; // down acts bf16 [NTOK][RANK]
// total ~13 MB

__device__ __forceinline__ int iclip(int v, int lo, int hi) {
    return v < lo ? lo : (v > hi ? hi : v);
}

// ---------------- 1) zero the per-expert counters ----------------
__global__ void zero_counts_kernel(int* counts) {
    if (threadIdx.x < 8) counts[threadIdx.x] = 0;
}

// ---------------- 2) fp32 -> bf16 weight conversion ----------------
__global__ __launch_bounds__(256)
void convert_weights_kernel(const float* __restrict__ Wd, const float* __restrict__ Wu,
                            __bf16* __restrict__ WdB, __bf16* __restrict__ WuB) {
    constexpr int N = NEXP * RANK * DIM;   // 917504, multiple of 1024
    int i = (blockIdx.x * 256 + threadIdx.x) * 4;
    if (i < N) {
        float4 v = *(const float4*)(Wd + i);
        WdB[i + 0] = (__bf16)v.x; WdB[i + 1] = (__bf16)v.y;
        WdB[i + 2] = (__bf16)v.z; WdB[i + 3] = (__bf16)v.w;
    } else {
        int j = i - N;
        if (j < N) {
            float4 v = *(const float4*)(Wu + j);
            WuB[j + 0] = (__bf16)v.x; WuB[j + 1] = (__bf16)v.y;
            WuB[j + 2] = (__bf16)v.z; WuB[j + 3] = (__bf16)v.w;
        }
    }
}

// ---------------- 3) bucket tokens per expert ----------------
__global__ __launch_bounds__(256)
void scatter_kernel(const int* __restrict__ ridx, const int* __restrict__ mask,
                    int* __restrict__ counts, int* __restrict__ perm) {
    int t = blockIdx.x * 256 + threadIdx.x;
    if (t >= NTOK) return;
    int k = ridx[t];
    if (k == FULL_KEY) return;
    bool act = false;
    #pragma unroll
    for (int i = 0; i < NEXP; ++i)
        if (iclip(mask[i], 0, FULL_KEY) == k) act = true;
    if (!act) return;
    int pos = atomicAdd(&counts[k], 1);
    perm[(size_t)k * CAP + pos] = t;
}

// ---------------- 6) inactive tokens: out = x ----------------
__global__ __launch_bounds__(256)
void copy_inactive_kernel(const float* __restrict__ x, const int* __restrict__ ridx,
                          const int* __restrict__ mask, float* __restrict__ out) {
    int t = blockIdx.x;
    int k = ridx[t];
    if (k != FULL_KEY) {
        bool act = false;
        #pragma unroll
        for (int i = 0; i < NEXP; ++i)
            if (iclip(mask[i], 0, FULL_KEY) == k) act = true;
        if (act) return;  // active tokens are written by up_kernel
    }
    const float4* xs = (const float4*)(x + (size_t)t * DIM);
    float4*       os = (float4*)(out + (size_t)t * DIM);
    os[threadIdx.x] = xs[threadIdx.x];   // 256 threads * float4 = 1024 floats
}

// ---------------- 4) down projection: down[t] = Wd[e] @ x[t] ----------------
// Per block: 32 gathered tokens (M=32), full RANK (N=128) across 8 waves, K=DIM.
// x staging is software-pipelined: next K-step's float4 is loaded into registers
// while the current step's WMMAs run, hiding the global-load latency.
__global__ __launch_bounds__(256)
void down_kernel(const float* __restrict__ x, const __bf16* __restrict__ WdB,
                 const int* __restrict__ counts, const int* __restrict__ perm,
                 __bf16* __restrict__ downB) {
    const int e   = blockIdx.y;
    const int m0  = blockIdx.x * 32;
    const int cnt = counts[e];
    if (m0 >= cnt) return;

    __shared__ int    toks[32];
    __shared__ __bf16 Atile[32][40];   // 32 tokens x 32 K (bf16), +8 pad vs bank conflicts

    const int tid = threadIdx.x;
    if (tid < 32) {
        int p = m0 + tid;
        toks[tid] = perm[(size_t)e * CAP + (p < cnt ? p : cnt - 1)];
    }
    __syncthreads();

    const int lane = tid & 31, wave = tid >> 5;
    const int mlo  = lane & 15, kh   = lane >> 4;
    const int n0   = wave * 16;                 // rank tile
    const __bf16* wd_e = WdB + (size_t)e * RANK * DIM;

    // hoisted per-thread staging pointer (constant row across the whole K loop)
    const int   srow = tid >> 3;
    const int   scol = (tid & 7) * 4;
    const float* xrow = x + (size_t)toks[srow] * DIM + scol;
    __bf16*      arow = &Atile[srow][scol];

    v8f c0 = {};  // token rows 0..15
    v8f c1 = {};  // token rows 16..31

    float4 v = *(const float4*)xrow;   // prologue load for k0 = 0

    for (int k0 = 0; k0 < DIM; k0 += 32) {
        // stage current K-step (Atile safe to overwrite: trailing barrier below)
        arow[0] = (__bf16)v.x; arow[1] = (__bf16)v.y;
        arow[2] = (__bf16)v.z; arow[3] = (__bf16)v.w;
        // prefetch next K-step before the barrier so it overlaps the WMMAs
        if (k0 + 32 < DIM) v = *(const float4*)(xrow + k0 + 32);
        __syncthreads();

        // A fragments (ISA 7.12.2: lanes 0-15 K=0..7 & 16..23, lanes 16-31 K=8..15 & 24..31)
        v16bf a0, a1, b;
        #pragma unroll
        for (int i = 0; i < 8; ++i) {
            a0[i]     = Atile[mlo][kh * 8 + i];
            a0[8 + i] = Atile[mlo][16 + kh * 8 + i];
            a1[i]     = Atile[16 + mlo][kh * 8 + i];
            a1[8 + i] = Atile[16 + mlo][16 + kh * 8 + i];
        }
        // B fragment: B[k][n] = Wd[e][n0+n][k0+k]; lane half gives K 8h..8h+7 & 16+8h..16+8h+7
        const __bf16* bp = wd_e + (size_t)(n0 + mlo) * DIM + k0 + kh * 8;
        #pragma unroll
        for (int j = 0; j < 16; ++j) b[j] = bp[(j & 7) + (j >> 3) * 16];

        c0 = __builtin_amdgcn_wmma_f32_16x16x32_bf16(false, a0, false, b, (short)0, c0, false, false);
        c1 = __builtin_amdgcn_wmma_f32_16x16x32_bf16(false, a1, false, b, (short)0, c1, false, false);
        __syncthreads();
    }

    // store down activations (bf16) by original token id; C layout: VGPR r -> M = kh*8+r, N = mlo.
    // Unconditional: clamped duplicate rows are the same token -> identical data (benign).
    __bf16* dcol = downB + n0 + mlo;
    #pragma unroll
    for (int r = 0; r < 8; ++r) {
        dcol[(size_t)toks[kh * 8 + r] * RANK]      = (__bf16)c0[r];
        dcol[(size_t)toks[16 + kh * 8 + r] * RANK] = (__bf16)c1[r];
    }
}

// ---------------- 5) up projection + residual: out[t] = x[t] + Wu[e] @ down[t] ----------------
// Per block: 32 gathered tokens (halves Wu L2 re-reads vs 16), DIM split into
// 8 waves * 128 cols, K=RANK=128 fully register-resident as 2x4 A fragments.
__global__ __launch_bounds__(256)
void up_kernel(const float* __restrict__ x, const __bf16* __restrict__ WuB,
               const int* __restrict__ counts, const int* __restrict__ perm,
               const __bf16* __restrict__ downB, float* __restrict__ out) {
    const int e   = blockIdx.y;
    const int m0  = blockIdx.x * 32;
    const int cnt = counts[e];
    if (m0 >= cnt) return;

    __shared__ int    toks[32];
    __shared__ __bf16 Atile[32][RANK + 8];   // 32 x 128 bf16, row stride 272B (16B aligned)

    const int tid = threadIdx.x;
    if (tid < 32) {
        int p = m0 + tid;
        toks[tid] = perm[(size_t)e * CAP + (p < cnt ? p : cnt - 1)];
    }
    __syncthreads();
    {   // stage full A: 32 rows x 128 bf16 = 4096 elems; 256 thr * 16 elems (2x uint4)
        int r = tid >> 3;
        int c = (tid & 7) * 16;
        const __bf16* src = downB + (size_t)toks[r] * RANK + c;
        *(uint4*)&Atile[r][c]     = *(const uint4*)src;
        *(uint4*)&Atile[r][c + 8] = *(const uint4*)(src + 8);
    }
    __syncthreads();

    const int lane = tid & 31, wave = tid >> 5;
    const int mlo  = lane & 15, kh   = lane >> 4;

    // preload all 4 K-step A fragments for both 16-token halves (K=128)
    v16bf afr0[4], afr1[4];
    #pragma unroll
    for (int ks = 0; ks < 4; ++ks)
        #pragma unroll
        for (int i = 0; i < 8; ++i) {
            afr0[ks][i]     = Atile[mlo][ks * 32 + kh * 8 + i];
            afr0[ks][8 + i] = Atile[mlo][ks * 32 + 16 + kh * 8 + i];
            afr1[ks][i]     = Atile[16 + mlo][ks * 32 + kh * 8 + i];
            afr1[ks][8 + i] = Atile[16 + mlo][ks * 32 + 16 + kh * 8 + i];
        }

    const __bf16* wu_e = WuB + (size_t)e * DIM * RANK;

    #pragma unroll
    for (int nt = 0; nt < 8; ++nt) {
        const int nbase = wave * 128 + nt * 16;   // output-dim tile
        v8f acc0 = {}, acc1 = {};
        #pragma unroll
        for (int ks = 0; ks < 4; ++ks) {
            v16bf b;
            // B[k][n] = Wu[e][nbase+n][k]; rows of Wu are contiguous in rank
            const __bf16* bp = wu_e + (size_t)(nbase + mlo) * RANK + ks * 32 + kh * 8;
            #pragma unroll
            for (int j = 0; j < 16; ++j) b[j] = bp[(j & 7) + (j >> 3) * 16];
            acc0 = __builtin_amdgcn_wmma_f32_16x16x32_bf16(false, afr0[ks], false, b,
                                                           (short)0, acc0, false, false);
            acc1 = __builtin_amdgcn_wmma_f32_16x16x32_bf16(false, afr1[ks], false, b,
                                                           (short)0, acc1, false, false);
        }
        // fused residual epilogue; unconditional (clamped duplicates write identical values)
        #pragma unroll
        for (int r = 0; r < 8; ++r) {
            size_t i0 = (size_t)toks[kh * 8 + r] * DIM + nbase + mlo;
            out[i0] = x[i0] + acc0[r];
            size_t i1 = (size_t)toks[16 + kh * 8 + r] * DIM + nbase + mlo;
            out[i1] = x[i1] + acc1[r];
        }
    }
}

// ---------------- launch ----------------
extern "C" void kernel_launch(void* const* d_in, const int* in_sizes, int n_in,
                              void* d_out, int out_size, void* d_ws, size_t ws_size,
                              hipStream_t stream) {
    const float* x    = (const float*)d_in[0];
    const int*   ridx = (const int*)d_in[1];
    const int*   mask = (const int*)d_in[2];
    const float* Wd   = (const float*)d_in[3];
    const float* Wu   = (const float*)d_in[4];
    float*       out  = (float*)d_out;

    char* ws = (char*)d_ws;
    int*    counts = (int*)(ws + OFF_COUNTS);
    int*    perm   = (int*)(ws + OFF_PERM);
    __bf16* WdB    = (__bf16*)(ws + OFF_WDB);
    __bf16* WuB    = (__bf16*)(ws + OFF_WUB);
    __bf16* downB  = (__bf16*)(ws + OFF_DOWN);

    zero_counts_kernel<<<1, 32, 0, stream>>>(counts);
    convert_weights_kernel<<<(2 * NEXP * RANK * DIM) / (256 * 4), 256, 0, stream>>>(Wd, Wu, WdB, WuB);
    scatter_kernel<<<NTOK / 256, 256, 0, stream>>>(ridx, mask, counts, perm);
    copy_inactive_kernel<<<NTOK, 256, 0, stream>>>(x, ridx, mask, out);
    down_kernel<<<dim3(CAP / 32, NEXP), 256, 0, stream>>>(x, WdB, counts, perm, downB);
    up_kernel<<<dim3(CAP / 32, NEXP), 256, 0, stream>>>(x, WuB, counts, perm, downB, out);
}
